// AttentionBase_11373073400356
// MI455X (gfx1250) — matrix operand order repaired and checked by reading, not compiled, not run on hardware
//
#include <hip/hip_runtime.h>

typedef _Float16 v16h __attribute__((ext_vector_type(16)));
typedef _Float16 v8h  __attribute__((ext_vector_type(8)));
typedef float    v8f  __attribute__((ext_vector_type(8)));

namespace {
constexpr int   kHeads = 16;
constexpr int   kSeq   = 2048;
constexpr int   kHd    = 64;
constexpr float kScale = 0.125f;
constexpr int   kBM    = 128;             // query rows per workgroup (8 waves x 16)
constexpr int   kBN    = 64;              // keys staged per iteration
constexpr int   kIters = kSeq / kBN;      // 32
constexpr int   kFragHalfs = 16 * 32 * 16;  // halfs per buffer (16 frags x 32 lanes x 16)
}

__global__ __launch_bounds__(256)
void fa_fwd_f16wmma(const float* __restrict__ Q,
                    const float* __restrict__ K,
                    const float* __restrict__ V,
                    float* __restrict__ Out) {
  // Double-buffered fragment store: 2 x (8 K-tiles as WMMA-A + 8 V-tiles as WMMA-B) = 32 KB
  __shared__ __align__(32) _Float16 sfrag[2 * kFragHalfs];

  const int tid  = threadIdx.x;
  const int lane = tid & 31;
  const int wave = tid >> 5;
  const int lh   = lane >> 4;   // half-wave selector
  const int l15  = lane & 15;

  const int b    = blockIdx.y;        // batch*head, 0..31
  const int m0   = blockIdx.x * kBM;  // query-row tile base
  const int b0   = b >> 4;
  const int head = b & 15;

  const size_t bbase = (size_t)b * kSeq * kHd;

  // ---- Q^T as WMMA-B fragments (two 32-wide c chunks), pre-scaled by 0.125 ----
  // B layout: lane holds column qrow=l&15; half h -> c = kc*32 + (l>>4)*8 + (h<8 ? h : h+8)
  v16h qB[2];
  {
    const float* qp = Q + bbase + (size_t)(m0 + wave * 16 + l15) * kHd + lh * 8;
#pragma unroll
    for (int kc = 0; kc < 2; ++kc) {
      const float4* q4 = (const float4*)(qp + kc * 32);
#pragma unroll
      for (int i = 0; i < 4; ++i) {
        float4 x = q4[i + (i >> 1) * 2];
        qB[kc][4 * i + 0] = (_Float16)(x.x * kScale);
        qB[kc][4 * i + 1] = (_Float16)(x.y * kScale);
        qB[kc][4 * i + 2] = (_Float16)(x.z * kScale);
        qB[kc][4 * i + 3] = (_Float16)(x.w * kScale);
      }
    }
  }

  // Loader role constants.
  // K loader: thread -> (key row jrel, 16-float column group cg)
  const int jrel  = tid & 63;
  const int cg    = tid >> 6;
  const int kfrag = (jrel >> 4) * 2 + (cg >> 1);
  const int khb   = (cg & 1) * 8;
  const int kln   = jrel & 15;
  // V loader: thread -> (column c, key 16-group)
  const int vc    = tid & 63;
  const int vr    = tid >> 6;
  const int vkc   = vr >> 1;
  const int vlh   = vr & 1;
  const int vfrag = 8 + (vc >> 4) * 2 + vkc;
  const int vln   = (vc & 15) + vlh * 16;

  // Load one 64-key block of K/V from global, converting f32 -> f16 into registers.
  auto load_block = [&](int j0, v8h& klo, v8h& khi, v8h& va, v8h& vb2) {
    const float4* kp = (const float4*)(K + bbase + (size_t)(j0 + jrel) * kHd + cg * 16);
    float4 x0 = kp[0], x1 = kp[1], x2 = kp[2], x3 = kp[3];
    klo[0] = (_Float16)x0.x; klo[1] = (_Float16)x0.y; klo[2] = (_Float16)x0.z; klo[3] = (_Float16)x0.w;
    klo[4] = (_Float16)x1.x; klo[5] = (_Float16)x1.y; klo[6] = (_Float16)x1.z; klo[7] = (_Float16)x1.w;
    khi[0] = (_Float16)x2.x; khi[1] = (_Float16)x2.y; khi[2] = (_Float16)x2.z; khi[3] = (_Float16)x2.w;
    khi[4] = (_Float16)x3.x; khi[5] = (_Float16)x3.y; khi[6] = (_Float16)x3.z; khi[7] = (_Float16)x3.w;
    const float* vp = V + bbase + (size_t)(j0 + vkc * 32 + vlh * 8) * kHd + vc;
#pragma unroll
    for (int h = 0; h < 8; ++h) va[h]  = (_Float16)vp[(size_t)h * kHd];
#pragma unroll
    for (int h = 0; h < 8; ++h) vb2[h] = (_Float16)vp[(size_t)(16 + h) * kHd];
  };

  // Store a staged block into LDS in fragment order.
  // K (WMMA-A): lane row = key = l&15; half h -> c = kc*32 + (l>>4)*8 + (h<8 ? h : h+8)
  // V (WMMA-B): lane col = c; half h -> k = kc*32 + (l>>4)*8 + (h<8 ? h : h+8)
  auto store_block = [&](int buf, const v8h& klo, const v8h& khi, const v8h& va, const v8h& vb2) {
    _Float16* base = sfrag + buf * kFragHalfs;
    *(v8h*)&base[(kfrag * 32 + kln) * 16 + khb]      = klo;
    *(v8h*)&base[(kfrag * 32 + kln + 16) * 16 + khb] = khi;
    *(v8h*)&base[(vfrag * 32 + vln) * 16]            = va;
    *(v8h*)&base[(vfrag * 32 + vln) * 16 + 8]        = vb2;
  };

  v8f oacc[4];
  {
    v8f zero = {};
#pragma unroll
    for (int ct = 0; ct < 4; ++ct) oacc[ct] = zero;
  }
  float mrun = -1e30f;
  float lrun = 0.0f;

  // Prologue: stage block 0 into buffer 0.
  {
    v8h klo, khi, va, vb2;
    load_block(0, klo, khi, va, vb2);
    store_block(0, klo, khi, va, vb2);
  }
  __syncthreads();

  for (int it = 0; it < kIters; ++it) {
    const int  cur     = it & 1;
    const bool hasNext = (it + 1) < kIters;

    // Issue next block's global loads early; latency hidden behind this iteration's WMMAs.
    v8h nklo, nkhi, nva, nvb;
    if (hasNext) load_block((it + 1) * kBN, nklo, nkhi, nva, nvb);

    const _Float16* fb = sfrag + cur * kFragHalfs;

    // ---- GEMM1: S^T(key, qrow) = K x Q^T; 4 tiles, chained over the 64-deep contraction ----
    v8f st[4];
#pragma unroll
    for (int jt = 0; jt < 4; ++jt) {
      v16h aK0 = *(const v16h*)&fb[((jt * 2 + 0) * 32 + lane) * 16];
      v16h aK1 = *(const v16h*)&fb[((jt * 2 + 1) * 32 + lane) * 16];
      v8f  z   = {};
      v8f  s0  = __builtin_amdgcn_wmma_f32_16x16x32_f16(false, aK0, false, qB[0],
                                                        (short)0, z, false, false);
      st[jt]   = __builtin_amdgcn_wmma_f32_16x16x32_f16(false, aK1, false, qB[1],
                                                        (short)0, s0, false, false);
    }

    // ---- online softmax over this 64-key block (qrow = lane&15 in S^T layout) ----
    float tmax = -1e30f;
#pragma unroll
    for (int jt = 0; jt < 4; ++jt)
#pragma unroll
      for (int v = 0; v < 8; ++v) tmax = fmaxf(tmax, st[jt][v]);
    tmax = fmaxf(tmax, __shfl_xor(tmax, 16, 32));  // combine the two key-octets
    const float mnew  = fmaxf(mrun, tmax);
    const float alpha = __expf(mrun - mnew);
    mrun = mnew;

    float rsum = 0.0f;
    v16h  aP[2];  // P as WMMA-A fragments, built fully in-lane from S^T D registers
#pragma unroll
    for (int jt = 0; jt < 4; ++jt)
#pragma unroll
      for (int v = 0; v < 8; ++v) {
        float pv = __expf(st[jt][v] - mnew);
        rsum += pv;
        aP[jt >> 1][(jt & 1) * 8 + v] = (_Float16)pv;
      }
    rsum += __shfl_xor(rsum, 16, 32);
    lrun = lrun * alpha + rsum;

    // rescale O accumulator: O's row for vgpr v is qrow = (lane>>4)*8 + v
#pragma unroll
    for (int v = 0; v < 8; ++v) {
      const float af = __shfl(alpha, (lh << 3) + v, 32);
#pragma unroll
      for (int ct = 0; ct < 4; ++ct) oacc[ct][v] *= af;
    }

    // ---- GEMM2: O += P x V ----
#pragma unroll
    for (int ct = 0; ct < 4; ++ct) {
      v16h bV0 = *(const v16h*)&fb[((8 + ct * 2 + 0) * 32 + lane) * 16];
      v16h bV1 = *(const v16h*)&fb[((8 + ct * 2 + 1) * 32 + lane) * 16];
      oacc[ct] = __builtin_amdgcn_wmma_f32_16x16x32_f16(false, aP[0], false, bV0,
                                                        (short)0, oacc[ct], false, false);
      oacc[ct] = __builtin_amdgcn_wmma_f32_16x16x32_f16(false, aP[1], false, bV1,
                                                        (short)0, oacc[ct], false, false);
    }

    // Stage next block into the alternate buffer; safe because buffer cur^1 was last
    // read before the previous barrier.
    if (hasNext) store_block(cur ^ 1, nklo, nkhi, nva, nvb);
    __syncthreads();  // single barrier per iteration
  }

  // ---- normalize and write Out[b0][row][head*64 + c] ----
  const float rinv = 1.0f / lrun;
  float rf[8];
#pragma unroll
  for (int v = 0; v < 8; ++v) rf[v] = __shfl(rinv, (lh << 3) + v, 32);

  float* ob = Out + (size_t)b0 * kSeq * (kHd * kHeads) + (size_t)head * kHd;
#pragma unroll
  for (int ct = 0; ct < 4; ++ct)
#pragma unroll
    for (int v = 0; v < 8; ++v) {
      const int    qr  = (lh << 3) + v;
      const size_t row = (size_t)(m0 + wave * 16 + qr);
      ob[row * (kHd * kHeads) + ct * 16 + l15] = oacc[ct][v] * rf[v];
    }
}

extern "C" void kernel_launch(void* const* d_in, const int* in_sizes, int n_in,
                              void* d_out, int out_size, void* d_ws, size_t ws_size,
                              hipStream_t stream) {
  (void)in_sizes; (void)n_in; (void)d_ws; (void)ws_size; (void)out_size;
  const float* Q = (const float*)d_in[0];
  const float* K = (const float*)d_in[1];
  const float* V = (const float*)d_in[2];
  float*       O = (float*)d_out;
  dim3 grid(kSeq / kBM, 32, 1);  // 16 row-tiles x 32 batch-heads
  dim3 block(256, 1, 1);         // 8 wave32
  hipLaunchKernelGGL(fa_fwd_f16wmma, grid, block, 0, stream, Q, K, V, O);
}
